// PT_37795712204979
// MI455X (gfx1250) — matrix operand order repaired and checked by reading, not compiled, not run on hardware
//
#include <hip/hip_runtime.h>
#include <hip/hip_bf16.h>
#include <math.h>

// ---------------- CDNA5 WMMA types ----------------
typedef __attribute__((ext_vector_type(16))) __bf16 v16bf;
typedef __attribute__((ext_vector_type(8)))  float  v8f;

union V16U { v16bf v; unsigned short u[16]; uint4 q[2]; };

__device__ __forceinline__ unsigned short f2bf(float f) {
    unsigned int u = __float_as_uint(f);
    u += 0x7FFFu + ((u >> 16) & 1u);           // round-to-nearest-even
    return (unsigned short)(u >> 16);
}
__device__ __forceinline__ float bf2f(unsigned short x) {
    return __uint_as_float(((unsigned int)x) << 16);
}

#define BDIM 256
static const int BN = 16 * 2048;

// ---------------- conv1 (k=1) + BN + ReLU -> x[B,N,64] f32 ----------------
__global__ void k_conv_bn_relu(const float* __restrict__ coords, const float* __restrict__ w,
                               const float* __restrict__ g, const float* __restrict__ b,
                               const float* __restrict__ m, const float* __restrict__ v,
                               float* __restrict__ out) {
    int idx = blockIdx.x * BDIM + threadIdx.x;           // over BN*64
    if (idx >= BN * 64) return;
    int o = idx & 63; int bn = idx >> 6;
    int n = bn & 2047; int bb = bn >> 11;
    const float* cb = coords + (size_t)bb * 3 * 2048 + n;
    float a = cb[0] * w[o*3+0] + cb[2048] * w[o*3+1] + cb[4096] * w[o*3+2];
    a = (a - m[o]) * rsqrtf(v[o] + 1e-5f) * g[o] + b[o];
    out[(size_t)bn * 64 + o] = fmaxf(a, 0.f);
}

// ---------------- f32 -> bf16 weight conversion ----------------
__global__ void k_f2bf(const float* __restrict__ s, unsigned short* __restrict__ d, int n) {
    int i = blockIdx.x * BDIM + threadIdx.x;
    if (i < n) d[i] = f2bf(s[i]);
}

// ---------------- SimpleRMSNorm (quirk: divide by sqrt(2e)) + bf16 cast ----------------
__global__ void k_srms_cast(const float* __restrict__ x, unsigned short* __restrict__ xn,
                            int e, float sqrt_d1) {
    int lane = threadIdx.x & 31;
    int wid = (blockIdx.x * BDIM + threadIdx.x) >> 5;
    if (wid >= BN) return;
    const float* row = x + (size_t)wid * e;
    float ss = 0.f;
    for (int i = lane; i < e; i += 32) { float t = row[i]; ss += t * t; }
    #pragma unroll
    for (int off = 16; off; off >>= 1) ss += __shfl_xor(ss, off, 32);
    float scale = 1.0f / (sqrtf(ss) / sqrt_d1 + 1e-8f);
    unsigned short* orow = xn + (size_t)wid * e;
    for (int i = lane; i < e; i += 32) orow[i] = f2bf(row[i] * scale);
}

// ---------------- RPE MLP -> reversed+transposed Toeplitz table ----------------
// tkr[c][s] = rpe(2047 - s), s in [0,4095)   (so rpe(i-j) = tkr[c][2047-i+j], contiguous in j)
__global__ void k_rpe(const float* __restrict__ Wp, const float* __restrict__ bp,
                      const float* __restrict__ Wl, const float* __restrict__ bl,
                      const float* __restrict__ Wo_, const float* __restrict__ bo_,
                      unsigned short* __restrict__ tkr, int rd, int d1) {
    int lane = threadIdx.x & 31, ws = threadIdx.x >> 5;
    int r = blockIdx.x * 8 + ws;                    // 0..4095, delta = r-2047
    float delta = (float)(r - 2047);
    __shared__ float sh[8][33];
    float srd = sqrtf((float)rd);

    float h = (lane < rd) ? (delta * Wp[lane] + bp[lane]) : 0.f;
    float ss = h * h;
    #pragma unroll
    for (int off = 16; off; off >>= 1) ss += __shfl_xor(ss, off, 32);
    float sc = 1.f / (sqrtf(ss) / srd + 1e-8f);
    sh[ws][lane] = fmaxf(h * sc, 0.f);
    __syncthreads();

    float h1 = 0.f;
    if (lane < rd) {
        for (int j = 0; j < rd; ++j) h1 += Wl[lane * rd + j] * sh[ws][j];
        h1 += bl[lane];
    }
    ss = h1 * h1;
    #pragma unroll
    for (int off = 16; off; off >>= 1) ss += __shfl_xor(ss, off, 32);
    sc = 1.f / (sqrtf(ss) / srd + 1e-8f);
    __syncthreads();
    sh[ws][lane] = fmaxf(h1 * sc, 0.f);
    __syncthreads();

    if (r < 4095) {                                  // s = 4094 - r in [0,4094]
        size_t s = (size_t)(4094 - r);
        for (int o = lane; o < d1; o += 32) {
            float acc = bo_[o];
            for (int j = 0; j < rd; ++j) acc += Wo_[o * rd + j] * sh[ws][j];
            tkr[(size_t)o * 4096 + s] = f2bf(acc);
        }
    }
}

// ---------------- WMMA GEMM: C[M,Nc] = A[M,K] * W[Nc,K]^T (+bias) ----------------
// one wave = one 16-row m-tile x four 16-col n-tiles (A fragment reused 4x).
// All fragment loads are issued before the wmma block so the backend can emit
// partial s_wait_loadcnt values (loads return in order) instead of full drains.
// mode 0: f32 row-major; mode 1: silu->bf16 row-major; mode 2: f32 transposed [B,Nc,2048];
// mode 3: silu->bf16 transposed [Nc,B,2048] (channel-major for the Toeplitz stage)
__global__ void k_gemm_wmma(const unsigned short* __restrict__ A, const unsigned short* __restrict__ W,
                            const float* __restrict__ bias, float* __restrict__ outF,
                            unsigned short* __restrict__ outB, int M, int Nc, int K, int mode) {
    int lane = threadIdx.x & 31;
    int wid = (blockIdx.x * BDIM + threadIdx.x) >> 5;
    int tm = M >> 4, tn4 = Nc >> 6;
    if (wid >= tm * tn4) return;
    int m0 = (wid % tm) << 4;
    int n0 = (wid / tm) << 6;                        // 4 n-tiles per wave

    v8f acc[4] = {};
    int arow = m0 + (lane & 15);
    int akb  = (lane < 16) ? 0 : 16;                 // A: element s holds k = 16*(lane/16) + s (contiguous order)
    int bkb  = (lane < 16) ? 0 : 16;                 // B: same contiguous k order as A
    const unsigned short* ap = A + (size_t)arow * K + akb;
    const unsigned short* wp0 = W + (size_t)(n0 + (lane & 15)) * K + bkb;

    for (int k0 = 0; k0 < K; k0 += 32) {
        V16U a, b[4];
        #pragma unroll
        for (int e2 = 0; e2 < 8; ++e2) { a.u[e2] = ap[k0 + e2]; a.u[8 + e2] = ap[k0 + 8 + e2]; }
        #pragma unroll
        for (int t = 0; t < 4; ++t) {
            const unsigned short* wp = wp0 + (size_t)(t << 4) * K + k0;  // 32B-aligned
            b[t].q[0] = ((const uint4*)wp)[0];
            b[t].q[1] = ((const uint4*)wp)[1];
        }
        #pragma unroll
        for (int t = 0; t < 4; ++t)
            acc[t] = __builtin_amdgcn_wmma_f32_16x16x32_bf16(false, a.v, false, b[t].v,
                                                             (short)0, acc[t], false, false);
    }

    int rbase = m0 + ((lane < 16) ? 0 : 8);
    #pragma unroll
    for (int t = 0; t < 4; ++t) {
        int n = n0 + (t << 4) + (lane & 15);
        float bv = bias ? bias[n] : 0.f;
        #pragma unroll
        for (int rr = 0; rr < 8; ++rr) {
            int m = rbase + rr;
            float val = acc[t][rr] + bv;
            if (mode == 0) {
                outF[(size_t)m * Nc + n] = val;
            } else if (mode == 1) {
                float s = val / (1.f + __expf(-val));             // SiLU
                outB[(size_t)m * Nc + n] = f2bf(s);
            } else if (mode == 2) {                               // feature: [B, Nc, 2048]
                int bb = m >> 11, i = m & 2047;
                outF[(((size_t)bb * Nc + n) << 11) + i] = val;
            } else {                                              // [Nc, B, 2048]
                int bb = m >> 11, i = m & 2047;
                float s = val / (1.f + __expf(-val));
                outB[((((size_t)n << 4) + bb) << 11) + i] = f2bf(s);
            }
        }
    }
}

// ---------------- per-channel Toeplitz GEMM + SiLU-gate fusion ----------------
// out[b,i,c] = u_t[c][b][i] * sum_j tkr[c][2047-i+j] * v_t[c][b][j]
// one wave = one channel x four 16-row i-tiles (B fragment reused 4x);
// loads hoisted ahead of the wmma block for partial-wait pipelining.
__global__ void k_toeplitz_gate(const unsigned short* __restrict__ tkr, const unsigned short* __restrict__ v_t,
                                const unsigned short* __restrict__ u_t, unsigned short* __restrict__ g, int d1) {
    int lane = threadIdx.x & 31;
    int wid = (blockIdx.x * BDIM + threadIdx.x) >> 5;
    if (wid >= (d1 << 5)) return;
    int c  = wid >> 5;
    int i0 = (wid & 31) << 6;                        // 4 i-tiles: i0 .. i0+63

    int r   = lane & 15;
    int akb = (lane < 16) ? 0 : 16;
    int bb  = lane & 15;                             // batch == WMMA N column (B=16 exactly)
    int bkb = (lane < 16) ? 0 : 16;

    // A strip base for tile t: tkr[c] + 2047 - (i0 + 16t + r) + j0 + akb  (contiguous in kk)
    const unsigned short* ta = tkr + (size_t)c * 4096 + (2047 - (i0 + r)) + akb;
    const unsigned short* vb = v_t + (((size_t)c << 4) + bb) * 2048 + bkb;

    v8f acc[4] = {};
    for (int j0 = 0; j0 < 2048; j0 += 32) {
        V16U b, a4[4];
        const unsigned short* vp = vb + j0;          // 32B-aligned
        b.q[0] = ((const uint4*)vp)[0];
        b.q[1] = ((const uint4*)vp)[1];
        #pragma unroll
        for (int t = 0; t < 4; ++t) {                // align-2 contiguous strips
            const unsigned short* tp = ta + j0 - (t << 4);
            __builtin_memcpy(a4[t].u,     tp,      16);
            __builtin_memcpy(a4[t].u + 8, tp + 8,  16);
        }
        __builtin_prefetch((const void*)(ta + j0 + 32), 0, 1);
        #pragma unroll
        for (int t = 0; t < 4; ++t)
            acc[t] = __builtin_amdgcn_wmma_f32_16x16x32_bf16(false, a4[t].v, false, b.v,
                                                             (short)0, acc[t], false, false);
    }

    int rb = (lane < 16) ? 0 : 8;
    const unsigned short* ub = u_t + ((((size_t)c << 4) + bb) << 11);
    #pragma unroll
    for (int t = 0; t < 4; ++t) {
        int ibase = i0 + (t << 4) + rb;
        uint4 uq = *(const uint4*)(ub + ibase);      // 8 contiguous bf16, 16B-aligned
        const unsigned short* us = (const unsigned short*)&uq;
        #pragma unroll
        for (int rr = 0; rr < 8; ++rr) {
            int ii = ibase + rr;
            g[((size_t)bb * 2048 + ii) * d1 + c] = f2bf(acc[t][rr] * bf2f(us[rr]));
        }
    }
}

// ---------------- maxpool over N (feature rows are contiguous) ----------------
__global__ void k_maxpool(const float* __restrict__ feat, float* __restrict__ pooled) {
    int lane = threadIdx.x & 31;
    int wid = (blockIdx.x * BDIM + threadIdx.x) >> 5;     // over 16*1024 rows
    if (wid >= 16 * 1024) return;
    const float* row = feat + (size_t)wid * 2048;
    float mx = -3.4e38f;
    for (int i = lane; i < 2048; i += 32) mx = fmaxf(mx, row[i]);
    #pragma unroll
    for (int off = 16; off; off >>= 1) mx = fmaxf(mx, __shfl_xor(mx, off, 32));
    if (lane == 0) pooled[wid] = mx;
}

// ---------------- head: lin1 + BN2 + ReLU, then lin2 + bias ----------------
__global__ void k_head1(const float* __restrict__ pooled, const float* __restrict__ w,
                        const float* __restrict__ g, const float* __restrict__ b,
                        const float* __restrict__ m, const float* __restrict__ vv,
                        float* __restrict__ h1) {
    int idx = blockIdx.x * BDIM + threadIdx.x;            // over 16*512
    if (idx >= 16 * 512) return;
    int j = idx & 511, bb = idx >> 9;
    float acc = 0.f;
    for (int k = 0; k < 1024; ++k) acc += pooled[bb * 1024 + k] * w[j * 1024 + k];
    acc = (acc - m[j]) * rsqrtf(vv[j] + 1e-5f) * g[j] + b[j];
    h1[idx] = fmaxf(acc, 0.f);
}
__global__ void k_head2(const float* __restrict__ h1, const float* __restrict__ w,
                        const float* __restrict__ b, float* __restrict__ out) {
    int idx = blockIdx.x * BDIM + threadIdx.x;            // over 16*256
    if (idx >= 16 * 256) return;
    int o = idx & 255, bb = idx >> 8;
    float acc = b[o];
    for (int j = 0; j < 512; ++j) acc += h1[bb * 512 + j] * w[o * 512 + j];
    out[idx] = acc;
}

// ---------------- orchestration ----------------
extern "C" void kernel_launch(void* const* d_in, const int* in_sizes, int n_in,
                              void* d_out, int out_size, void* d_ws, size_t ws_size,
                              hipStream_t stream) {
    (void)in_sizes; (void)n_in; (void)out_size; (void)ws_size;
    const float* coords = (const float*)d_in[0];

    // workspace carving
    char* wp_ = (char*)d_ws;
    auto alloc = [&](size_t bytes) { void* p = (void*)wp_; wp_ += (bytes + 255) & ~(size_t)255; return p; };
    float*          xA   = (float*)alloc((size_t)BN * 512 * 4);
    float*          xB   = (float*)alloc((size_t)BN * 512 * 4);
    unsigned short* xn   = (unsigned short*)alloc((size_t)BN * 512 * 2);
    unsigned short* ubuf = (unsigned short*)alloc((size_t)BN * 1024 * 2);   // u_t [d1,16,2048]
    unsigned short* vbuf = (unsigned short*)alloc((size_t)BN * 1024 * 2);   // v_t [d1,16,2048]
    unsigned short* gbuf = (unsigned short*)alloc((size_t)BN * 1024 * 2);   // g   [B,N,d1]
    unsigned short* tkb  = (unsigned short*)alloc((size_t)1024 * 4096 * 2); // tkr [d1,4096]
    unsigned short* wWu  = (unsigned short*)alloc((size_t)1024 * 512 * 2);
    unsigned short* wWv  = (unsigned short*)alloc((size_t)1024 * 512 * 2);
    unsigned short* wWo  = (unsigned short*)alloc((size_t)1024 * 1024 * 2);
    float*          pooled = (float*)alloc(16 * 1024 * 4);
    float*          h1   = (float*)alloc(16 * 512 * 4);

    // conv1 + bn1 + relu -> xA [B,N,64]
    k_conv_bn_relu<<<(BN * 64 + BDIM - 1) / BDIM, BDIM, 0, stream>>>(
        coords, (const float*)d_in[1], (const float*)d_in[2], (const float*)d_in[3],
        (const float*)d_in[4], (const float*)d_in[5], xA);

    struct Cfg { int e, d1, rd, base; };
    Cfg cfg[4] = { {64,128,8,6}, {128,256,16,18}, {256,512,32,30}, {512,1024,32,42} };
    float* feat = (float*)d_out + 4096;               // feature region [16,1024,2048]

    float* xin = xA; float* xout = xB;
    for (int L = 0; L < 4; ++L) {
        const Cfg c = cfg[L];
        const float* Wu = (const float*)d_in[c.base + 0];
        const float* bu = (const float*)d_in[c.base + 1];
        const float* Wv = (const float*)d_in[c.base + 2];
        const float* bv = (const float*)d_in[c.base + 3];
        const float* Wo = (const float*)d_in[c.base + 4];
        const float* bo = (const float*)d_in[c.base + 5];
        const float* Wpos = (const float*)d_in[c.base + 6];
        const float* bpos = (const float*)d_in[c.base + 7];
        const float* Wl0  = (const float*)d_in[c.base + 8];
        const float* bl0  = (const float*)d_in[c.base + 9];
        const float* Wout = (const float*)d_in[c.base + 10];
        const float* bout = (const float*)d_in[c.base + 11];

        int nwu = c.d1 * c.e, nwo = c.d1 * c.d1;
        k_f2bf<<<(nwu + BDIM - 1) / BDIM, BDIM, 0, stream>>>(Wu, wWu, nwu);
        k_f2bf<<<(nwu + BDIM - 1) / BDIM, BDIM, 0, stream>>>(Wv, wWv, nwu);
        k_f2bf<<<(nwo + BDIM - 1) / BDIM, BDIM, 0, stream>>>(Wo, wWo, nwo);

        // reversed+transposed Toeplitz table tkr[d1, 4096]
        k_rpe<<<512, BDIM, 0, stream>>>(Wpos, bpos, Wl0, bl0, Wout, bout, tkb, c.rd, c.d1);

        // xn = srms(x) (divisor uses sqrt(2e) per the reference quirk)
        k_srms_cast<<<(BN * 32 + BDIM - 1) / BDIM, BDIM, 0, stream>>>(xin, xn, c.e, sqrtf(2.f * c.e));

        // u = silu(xn Wu^T + bu), v = silu(xn Wv^T + bv): both channel-major bf16
        int waves = (BN / 16) * (c.d1 / 64);
        int grid  = (waves + 7) / 8;
        k_gemm_wmma<<<grid, BDIM, 0, stream>>>(xn, wWu, bu, nullptr, ubuf, BN, c.d1, c.e, 3);
        k_gemm_wmma<<<grid, BDIM, 0, stream>>>(xn, wWv, bv, nullptr, vbuf, BN, c.d1, c.e, 3);

        // per-channel Toeplitz matmul fused with u-gating -> gbuf bf16 [B,N,d1]
        int twaves = c.d1 << 5;
        k_toeplitz_gate<<<(twaves + 7) / 8, BDIM, 0, stream>>>(tkb, vbuf, ubuf, gbuf, c.d1);

        // Wo projection (+bias). Layer 4 stores transposed directly into feature output.
        if (L < 3)
            k_gemm_wmma<<<grid, BDIM, 0, stream>>>(gbuf, wWo, bo, xout, nullptr, BN, c.d1, c.d1, 0);
        else
            k_gemm_wmma<<<grid, BDIM, 0, stream>>>(gbuf, wWo, bo, feat, nullptr, BN, 1024, 1024, 2);

        float* t = xin; xin = xout; xout = t;
    }

    // maxpool over N, then head
    k_maxpool<<<(16 * 1024) / 8, BDIM, 0, stream>>>(feat, pooled);
    k_head1<<<(16 * 512 + BDIM - 1) / BDIM, BDIM, 0, stream>>>(
        pooled, (const float*)d_in[54], (const float*)d_in[55], (const float*)d_in[56],
        (const float*)d_in[57], (const float*)d_in[58], h1);
    k_head2<<<(16 * 256 + BDIM - 1) / BDIM, BDIM, 0, stream>>>(
        h1, (const float*)d_in[59], (const float*)d_in[60], (float*)d_out);
}
